// JCDD_S_layer_23794118820498
// MI455X (gfx1250) — matrix operand order repaired and checked by reading, not compiled, not run on hardware
//
#include <hip/hip_runtime.h>
#include <cstdint>

// ---------------------------------------------------------------------------
// JCDD-S layer for MI455X (gfx1250). B=512, Nr=64, Nt=8, Ld=128, Lp=16,
// M=1024, LDPC_N=2048. Strategy: bf16 WMMA (f32 accum) for the three large
// GEMMs (UrT@YYp per batch, A^T@v batched, A@u batched); VALU for tiny-K ops.
// ---------------------------------------------------------------------------

typedef __attribute__((ext_vector_type(16))) __bf16 v16bf;
typedef __attribute__((ext_vector_type(8)))  __bf16 v8bf;
typedef __attribute__((ext_vector_type(8)))  float  v8f;
typedef __attribute__((ext_vector_type(4)))  int    v4i;

#if defined(__has_builtin)
#if __has_builtin(__builtin_amdgcn_global_load_async_to_lds_b128)
#define ATHENA_HAS_ASYNC 1
#endif
#endif

__device__ __forceinline__ void async_cp16(const float* g, float* l) {
#ifdef ATHENA_HAS_ASYNC
  __builtin_amdgcn_global_load_async_to_lds_b128(
      (__attribute__((address_space(1))) v4i*)(v4i*)(float*)g,
      (__attribute__((address_space(3))) v4i*)(v4i*)l, 0, 0);
#else
  *(float4*)l = *(const float4*)g;
#endif
}

__device__ __forceinline__ void async_wait() {
#ifdef ATHENA_HAS_ASYNC
#if __has_builtin(__builtin_amdgcn_s_wait_asynccnt)
  __builtin_amdgcn_s_wait_asynccnt(0);
#else
  asm volatile("s_wait_asynccnt 0" ::: "memory");
#endif
#endif
}

__device__ __forceinline__ v16bf ld_bf16x16(const __bf16* p0, const __bf16* p1) {
  union { v8bf h[2]; v16bf v; } u;
  u.h[0] = *(const v8bf*)p0;
  u.h[1] = *(const v8bf*)p1;
  return u.v;
}

__device__ __forceinline__ v8f wmma_bf16f32(v16bf a, v16bf b, v8f c) {
  // 8 args: (neg_a, A, neg_b, B, c_mod, C, reuse_a, reuse_b)
  return __builtin_amdgcn_wmma_f32_16x16x32_bf16(false, a, false, b, (short)0, c,
                                                 false, false);
}

// ---------------------------------------------------------------------------
// Prep: build UrT (128x128), Ur (128x128) f32 + UrT bf16 in workspace.
// UrT = [[rUr^T, iUr^T],[-iUr^T, rUr^T]];  Ur = [[rUr, -iUr],[iUr, rUr]]
// ---------------------------------------------------------------------------
__global__ __launch_bounds__(256) void k_prep(const float* __restrict__ rUr,
                                              const float* __restrict__ iUr,
                                              float* __restrict__ UrT,
                                              float* __restrict__ Ur,
                                              __bf16* __restrict__ UrTb) {
  int e = blockIdx.x * 256 + threadIdx.x;
  if (e >= 16384) return;
  int i = e >> 7, j = e & 127;
  float vT, vU;
  if (i < 64) {
    vT = (j < 64) ? rUr[j * 64 + i] : iUr[(j - 64) * 64 + i];
    vU = (j < 64) ? rUr[i * 64 + j] : -iUr[i * 64 + (j - 64)];
  } else {
    vT = (j < 64) ? -iUr[j * 64 + (i - 64)] : rUr[(j - 64) * 64 + (i - 64)];
    vU = (j < 64) ? iUr[(i - 64) * 64 + j] : rUr[(i - 64) * 64 + (j - 64)];
  }
  UrT[e] = vT;
  UrTb[e] = (__bf16)vT;
  Ur[e] = vU;
}

// A (1024x2048) -> bf16 row-major + bf16 transpose (2048x1024)
__global__ __launch_bounds__(256) void k_convA(const float* __restrict__ A,
                                               __bf16* __restrict__ Ab,
                                               __bf16* __restrict__ Atb) {
  int e = blockIdx.x * 256 + threadIdx.x;
  if (e >= 1024 * 2048) return;
  int m = e >> 11, k = e & 2047;
  float v = A[e];
  Ab[e] = (__bf16)v;
  Atb[(size_t)k * 1024 + m] = (__bf16)v;
}

// V[b][k] = theta[k] - z[b][k] - lambda1[b][k]  (bf16, batch-major = B-operand)
__global__ __launch_bounds__(256) void k_V(const float* __restrict__ theta,
                                           const float* __restrict__ z,
                                           const float* __restrict__ l1,
                                           __bf16* __restrict__ Vb) {
  int e = blockIdx.x * 256 + threadIdx.x;
  if (e >= 512 * 1024) return;
  int k = e & 1023;
  Vb[e] = (__bf16)(theta[k] - z[e] - l1[e]);
}

// ---------------------------------------------------------------------------
// K1: per-batch channel estimation + detection front end. 1 block per b.
// ---------------------------------------------------------------------------
struct alignas(16) UnionA {
  union {
    __bf16 bt[144 * 128];                 // YYp^T bf16 (phases 0-5)
    struct { float x[16 * 128]; float f[16 * 128]; } xf;  // X (async) + F1
  };
};

__global__ __launch_bounds__(256) void k1_channel(
    const float* __restrict__ sigma2, const float* __restrict__ Xg,
    const float* __restrict__ XXpg,   const float* __restrict__ Yg,
    const float* __restrict__ YYpg,   const float* __restrict__ Hg,
    const float* __restrict__ lamWg,  const float* __restrict__ rUt,
    const float* __restrict__ iUt,    const float* __restrict__ gUrT,
    const float* __restrict__ gUr,    const __bf16* __restrict__ gUrTb,
    const float* __restrict__ epsP,   const float* __restrict__ taoP,
    const float* __restrict__ facP,   float* __restrict__ outH,
    float* __restrict__ outLam,       float* __restrict__ fbres) {
  __shared__ UnionA uA;                               // 36864 B
  __shared__ alignas(16) float sDelta[128 * 144];     // 73728 B
  __shared__ alignas(16) float sM16[128 * 16];        // UrH(blkb)->UrHv(blkb)->W
  __shared__ float sV8a[128 * 8];                     // H in -> Hvest
  __shared__ float sV8b[128 * 8];                     // Hv_temp -> H_new
  __shared__ float sUtTX[16 * 144];                   // UtT @ XXp
  __shared__ float sUtT[16 * 16];
  __shared__ float sUt[16 * 8];
  __shared__ float sUtT2[16 * 8];
  __shared__ float sWTW[16 * 16];

  const int b = blockIdx.x;
  const int tid = threadIdx.x;
  const float taov = taoP[0];
  const float lam = facP[0] * lamWg[b];

  // ---- phase 0: stage small operands ----
  {
    int i = tid >> 4, j = tid & 15;  // UtT = [[rUt^T, iUt^T],[-iUt^T, rUt^T]]
    float v;
    if (i < 8) v = (j < 8) ? rUt[j * 8 + i] : iUt[(j - 8) * 8 + i];
    else       v = (j < 8) ? -iUt[j * 8 + (i - 8)] : rUt[(j - 8) * 8 + (i - 8)];
    sUtT[tid] = v;
  }
  if (tid < 128) {
    int i = tid >> 3, t = tid & 7;
    sUt[tid]   = (i < 8) ? rUt[i * 8 + t] : iUt[(i - 8) * 8 + t];   // [rUt; iUt]
    sUtT2[tid] = (i < 8) ? rUt[t * 8 + i] : -iUt[t * 8 + (i - 8)];  // [rUt^T; -iUt^T]
  }
  for (int e = tid; e < 1024; e += 256) sV8a[e] = Hg[(size_t)b * 1024 + e];
  for (int e = tid; e < 18432; e += 256) {  // YYp staged transposed bf16
    int k = e / 144, n = e - k * 144;
    uA.bt[n * 128 + k] = (__bf16)YYpg[(size_t)b * 18432 + e];
  }
  __syncthreads();

  // ---- phase 1: UrH_temp = UrT @ H  -> blkb into sM16 (128x16) ----
  for (int e = tid; e < 1024; e += 256) {
    int m = e >> 3, t = e & 7;
    float acc = 0.f;
    const float* ur = &gUrT[m * 128];
    for (int k = 0; k < 128; ++k) acc += ur[k] * sV8a[k * 8 + t];
    if (m < 64) { sM16[m * 16 + t] = acc; sM16[(m + 64) * 16 + 8 + t] = acc; }
    else        { sM16[(m - 64) * 16 + 8 + t] = -acc; sM16[m * 16 + t] = acc; }
  }
  __syncthreads();

  // ---- phase 2: Hv_temp = UrH @ Ut (128x8);  phase 3: UtTXXp = UtT @ XXp ----
  for (int e = tid; e < 1024; e += 256) {
    int m = e >> 3, t = e & 7;
    float acc = 0.f;
    for (int j = 0; j < 16; ++j) acc += sM16[m * 16 + j] * sUt[j * 8 + t];
    sV8b[e] = acc;
  }
  for (int e = tid; e < 2304; e += 256) {
    int i = e / 144, l = e - i * 144;
    float acc = 0.f;
    for (int j = 0; j < 16; ++j)
      acc += sUtT[i * 16 + j] * XXpg[(size_t)b * 2304 + j * 144 + l];
    sUtTX[e] = acc;
  }
  __syncthreads();

  // ---- phase 4: T1 = blkb(Hv_temp) @ UtTXXp  (128x144) ----
  for (int e = tid; e < 18432; e += 256) {
    int m = e / 144, l = e - m * 144;
    float acc = 0.f;
    for (int j = 0; j < 8; ++j) acc += sV8b[m * 8 + j] * sUtTX[j * 144 + l];
    if (m < 64)
      for (int j = 0; j < 8; ++j) acc -= sV8b[(m + 64) * 8 + j] * sUtTX[(j + 8) * 144 + l];
    else
      for (int j = 0; j < 8; ++j) acc += sV8b[(m - 64) * 8 + j] * sUtTX[(j + 8) * 144 + l];
    sDelta[e] = acc;
  }
  __syncthreads();

  // ---- phase 5: delta_temp = T1 - UrT@YYp via WMMA bf16 (128x128x144) ----
  {
    const int wave = tid >> 5, lane = tid & 31;
    const int m = wave * 16 + (lane & 15);
    const int kg = (lane >> 4) * 8;
    const int nl = lane & 15;
    const int kb = (lane >> 4) * 16;
    const __bf16* arow = gUrTb + (size_t)m * 128;
    for (int nt = 0; nt < 9; ++nt) {
      v8f acc = {0.f, 0.f, 0.f, 0.f, 0.f, 0.f, 0.f, 0.f};
      const int n = nt * 16 + nl;
      const __bf16* brow = &uA.bt[n * 128];
#pragma unroll
      for (int k0 = 0; k0 < 128; k0 += 32) {
        v16bf a  = ld_bf16x16(arow + k0 + kg, arow + k0 + kg + 16);
        v16bf bm = ld_bf16x16(brow + k0 + kg, brow + k0 + kg + 16);
        acc = wmma_bf16f32(a, bm, acc);
      }
      const int mh = wave * 16 + (lane >> 4) * 8;
#pragma unroll
      for (int r = 0; r < 8; ++r) sDelta[(mh + r) * 144 + n] -= acc[r];
    }
  }
  __syncthreads();

  // async-stage X tile (overlaps phases 6-11a); bt region is dead now
  for (int c = tid; c < 512; c += 256)
    async_cp16(&Xg[(size_t)b * 2048 + c * 4], &uA.xf.x[c * 4]);

  // ---- phase 6: Hvest = Hv_temp - tao * blkb(delta) @ UtTXXp^T ----
  for (int e = tid; e < 1024; e += 256) {
    int m = e >> 3, t = e & 7;
    float s = 0.f;
    if (m < 64) {
      for (int l = 0; l < 144; ++l)
        s += sDelta[m * 144 + l] * sUtTX[t * 144 + l] +
             sDelta[(m + 64) * 144 + l] * sUtTX[(8 + t) * 144 + l];
    } else {
      int mi = m - 64;
      for (int l = 0; l < 144; ++l)
        s += sDelta[m * 144 + l] * sUtTX[t * 144 + l] -
             sDelta[mi * 144 + l] * sUtTX[(8 + t) * 144 + l];
    }
    sV8a[e] = sV8b[e] - taov * s;
  }
  __syncthreads();

  // ---- phase 7: complex magnitude shrinkage ----
  {
    const float thr = taov * epsP[0] * sigma2[b];
    for (int e = tid; e < 512; e += 256) {
      int m = e >> 3, t = e & 7;
      float re = sV8a[m * 8 + t], im = sV8a[(m + 64) * 8 + t];
      float mag = sqrtf(re * re + im * im);
      float sc = (mag > thr) ? (mag - thr) / mag : 0.f;
      sV8a[m * 8 + t] = re * sc;
      sV8a[(m + 64) * 8 + t] = im * sc;
    }
  }
  __syncthreads();

  // ---- phase 8: UrHv = Ur @ Hvest -> blkb into sM16 ----
  for (int e = tid; e < 1024; e += 256) {
    int m = e >> 3, t = e & 7;
    float acc = 0.f;
    const float* ur = &gUr[m * 128];
    for (int k = 0; k < 128; ++k) acc += ur[k] * sV8a[k * 8 + t];
    if (m < 64) { sM16[m * 16 + t] = acc; sM16[(m + 64) * 16 + 8 + t] = acc; }
    else        { sM16[(m - 64) * 16 + 8 + t] = -acc; sM16[m * 16 + t] = acc; }
  }
  __syncthreads();

  // ---- phase 9: H_new = blkb(UrHv) @ UtT2 ----
  for (int e = tid; e < 1024; e += 256) {
    int m = e >> 3, t = e & 7;
    float acc = 0.f;
    for (int j = 0; j < 16; ++j) acc += sM16[m * 16 + j] * sUtT2[j * 8 + t];
    sV8b[e] = acc;
    outH[(size_t)b * 1024 + e] = acc;
  }
  __syncthreads();

  // ---- phase 10: W = blkb(H_new) (128x16) into sM16 ----
  for (int e = tid; e < 2048; e += 256) {
    int m = e >> 4, j = e & 15;
    float v;
    if (j < 8) v = sV8b[m * 8 + j];
    else v = (m < 64) ? -sV8b[(m + 64) * 8 + (j - 8)] : sV8b[(m - 64) * 8 + (j - 8)];
    sM16[e] = v;
  }
  __syncthreads();

  // ---- phase 11a: WTW (16x16) and F1 = W^T @ Y (16x128) ----
  {
    int i = tid >> 4, j = tid & 15;
    float acc = 0.f;
    for (int r = 0; r < 128; ++r) acc += sM16[r * 16 + i] * sM16[r * 16 + j];
    sWTW[tid] = acc;
  }
  {
    const int half = tid >> 7, l = tid & 127;
    float acc[8];
#pragma unroll
    for (int i = 0; i < 8; ++i) acc[i] = 0.f;
    for (int r = 0; r < 128; ++r) {
      float y = Yg[(size_t)b * 16384 + r * 128 + l];
#pragma unroll
      for (int i = 0; i < 8; ++i) acc[i] += sM16[r * 16 + half * 8 + i] * y;
    }
#pragma unroll
    for (int i = 0; i < 8; ++i) uA.xf.f[(half * 8 + i) * 128 + l] = acc[i];
  }
  async_wait();
  __syncthreads();

  // ---- phase 11b: F = F1 + lam*X - WTW@X ; fb_res = 2lam - 2*sqrt2*res ----
  for (int e = tid; e < 2048; e += 256) {
    int i = e >> 7, l = e & 127;
    float wx = 0.f;
    for (int j = 0; j < 16; ++j) wx += sWTW[i * 16 + j] * uA.xf.x[j * 128 + l];
    float f = uA.xf.f[e] + lam * uA.xf.x[e] - wx;
    int c = i >> 3, t = i & 7;
    fbres[(size_t)b * 2048 + l * 16 + t * 2 + c] = 2.f * lam - 2.8284271247461903f * f;
  }
  if (tid == 0) outLam[b] = lam;
}

// ---------------------------------------------------------------------------
// K2: Q = A^T(2048x1024) @ V(1024x512) via WMMA; u = clamp01((rho*Q+fbres-k)*w)
// ---------------------------------------------------------------------------
__global__ __launch_bounds__(256) void k2_u(
    const __bf16* __restrict__ Atb, const __bf16* __restrict__ Vb,
    const float* __restrict__ fbres, const float* __restrict__ LamA,
    const float* __restrict__ lamWg, const float* __restrict__ facP,
    const float* __restrict__ rhoP, const float* __restrict__ kapP,
    float* __restrict__ outU, __bf16* __restrict__ ubf) {
  const int lane = threadIdx.x & 31, wave = threadIdx.x >> 5;
  const int tile = blockIdx.x * 8 + wave;   // 4096 tiles: 128 (M) x 32 (N)
  const int tm = tile >> 5, tn = tile & 31;
  const int mA = tm * 16 + (lane & 15);
  const int kg = (lane >> 4) * 8;
  const int nB = tn * 16 + (lane & 15);
  const int kb = (lane >> 4) * 16;
  v8f acc = {0.f, 0.f, 0.f, 0.f, 0.f, 0.f, 0.f, 0.f};
  const __bf16* arow = Atb + (size_t)mA * 1024;
  const __bf16* brow = Vb + (size_t)nB * 1024;
  for (int k0 = 0; k0 < 1024; k0 += 32) {
    v16bf a  = ld_bf16x16(arow + k0 + kg, arow + k0 + kg + 16);
    v16bf bm = ld_bf16x16(brow + k0 + kg, brow + k0 + kg + 16);
    acc = wmma_bf16f32(a, bm, acc);
  }
  const float rh = rhoP[0], kp = kapP[0], fc = facP[0];
  const float lamb = fc * lamWg[nB];
  const int mh = tm * 16 + (lane >> 4) * 8;
#pragma unroll
  for (int r = 0; r < 8; ++r) {
    int m = mh + r;
    float q = rh * acc[r] + fbres[(size_t)nB * 2048 + m] - kp;
    float wv = 1.f / (rh * LamA[m] + 4.f * lamb - 2.f * kp);
    float un = q * wv;
    un = fminf(fmaxf(un, 0.f), 1.f);
    outU[(size_t)nB * 2048 + m] = un;
    ubf[(size_t)nB * 2048 + m] = (__bf16)un;
  }
}

// ---------------------------------------------------------------------------
// K3: R = A(1024x2048) @ u(2048x512) via WMMA; z / lambda1 updates
// ---------------------------------------------------------------------------
__global__ __launch_bounds__(256) void k3_z(
    const __bf16* __restrict__ Ab, const __bf16* __restrict__ ubf,
    const float* __restrict__ theta, const float* __restrict__ zg,
    const float* __restrict__ l1g, const float* __restrict__ relaxP,
    const float* __restrict__ accP, float* __restrict__ outZ,
    float* __restrict__ outL) {
  const int lane = threadIdx.x & 31, wave = threadIdx.x >> 5;
  const int tile = blockIdx.x * 8 + wave;   // 2048 tiles: 64 (M) x 32 (N)
  const int tm = tile >> 5, tn = tile & 31;
  const int mA = tm * 16 + (lane & 15);
  const int kg = (lane >> 4) * 8;
  const int nB = tn * 16 + (lane & 15);
  const int kb = (lane >> 4) * 16;
  v8f acc = {0.f, 0.f, 0.f, 0.f, 0.f, 0.f, 0.f, 0.f};
  const __bf16* arow = Ab + (size_t)mA * 2048;
  const __bf16* brow = ubf + (size_t)nB * 2048;
  for (int k0 = 0; k0 < 2048; k0 += 32) {
    v16bf a  = ld_bf16x16(arow + k0 + kg, arow + k0 + kg + 16);
    v16bf bm = ld_bf16x16(brow + k0 + kg, brow + k0 + kg + 16);
    acc = wmma_bf16f32(a, bm, acc);
  }
  const float rl = relaxP[0], ac = accP[0];
  const int mh = tm * 16 + (lane >> 4) * 8;
#pragma unroll
  for (int r = 0; r < 8; ++r) {
    int m = mh + r;
    float th = theta[m];
    float zo = zg[(size_t)nB * 1024 + m], lo = l1g[(size_t)nB * 1024 + m];
    float zt = th - rl * acc[r] - (1.f - rl) * (th - zo) - lo;
    float zn = fmaxf(zt, 0.f);
    float ln = zn - zt;
    outZ[(size_t)nB * 1024 + m] = zn + ac * (zn - zo);
    outL[(size_t)nB * 1024 + m] = ln + ac * (ln - lo);
  }
}

// ---------------------------------------------------------------------------
// K4: X_new / XXp_new from u_new; also copy acc_new passthrough
// ---------------------------------------------------------------------------
__global__ __launch_bounds__(256) void k4_x(
    const float* __restrict__ u, const float* __restrict__ rXp,
    const float* __restrict__ iXp, const float* __restrict__ accNew,
    float* __restrict__ outX, float* __restrict__ outXXp,
    float* __restrict__ outAcc) {
  int e = blockIdx.x * 256 + threadIdx.x;
  if (e == 0) outAcc[0] = accNew[0];
  if (e >= 512 * 16 * 144) return;
  int l = e % 144;
  int i = (e / 144) & 15;
  int b = e / (144 * 16);
  if (l < 128) {
    int c = i >> 3, t = i & 7;
    float uv = u[(size_t)b * 2048 + l * 16 + t * 2 + c];
    float v = (1.f - 2.f * uv) * 0.7071067811865476f;
    outXXp[e] = v;
    outX[(size_t)b * 2048 + i * 128 + l] = v;
  } else {
    int t = i & 7, lp = l - 128;
    outXXp[e] = (i < 8 ? rXp : iXp)[(size_t)b * 128 + t * 16 + lp];
  }
}

// ---------------------------------------------------------------------------
extern "C" void kernel_launch(void* const* d_in, const int* in_sizes, int n_in,
                              void* d_out, int out_size, void* d_ws,
                              size_t ws_size, hipStream_t stream) {
  const float* sigma2 = (const float*)d_in[0];
  const float* X      = (const float*)d_in[1];
  const float* XXp    = (const float*)d_in[2];
  const float* Y      = (const float*)d_in[3];
  const float* YYp    = (const float*)d_in[4];
  const float* rXp    = (const float*)d_in[5];
  const float* iXp    = (const float*)d_in[6];
  const float* H      = (const float*)d_in[7];
  const float* lamW   = (const float*)d_in[8];
  const float* z      = (const float*)d_in[10];
  const float* l1     = (const float*)d_in[11];
  const float* accNew = (const float*)d_in[12];
  const float* A      = (const float*)d_in[13];
  const float* LamA   = (const float*)d_in[14];
  const float* theta  = (const float*)d_in[15];
  const float* rUr    = (const float*)d_in[16];
  const float* iUr    = (const float*)d_in[17];
  const float* rUt    = (const float*)d_in[18];
  const float* iUt    = (const float*)d_in[19];
  const float* rho    = (const float*)d_in[20];
  const float* kappa  = (const float*)d_in[21];
  const float* eps    = (const float*)d_in[22];
  const float* tao    = (const float*)d_in[23];
  const float* factor = (const float*)d_in[24];
  const float* relax  = (const float*)d_in[25];
  const float* accP   = (const float*)d_in[26];

  float* out = (float*)d_out;
  float* outX   = out;                 // (512,16,128)
  float* outXXp = out + 1048576;       // (512,16,144)
  float* outH   = out + 2228224;       // (512,128,8)
  float* outLam = out + 2752512;       // (512)
  float* outU   = out + 2753024;       // (512,2048)
  float* outZ   = out + 3801600;       // (512,1024)
  float* outL   = out + 4325888;       // (512,1024)
  float* outAcc = out + 4850176;       // (1)

  char* w = (char*)d_ws;               // ~15.2 MB of workspace used
  float*  UrT   = (float*)(w + 0);          // 128*128 f32
  float*  Ur    = (float*)(w + 65536);      // 128*128 f32
  __bf16* UrTb  = (__bf16*)(w + 131072);    // 128*128 bf16
  __bf16* Ab    = (__bf16*)(w + 163840);    // 1024*2048 bf16
  __bf16* Atb   = (__bf16*)(w + 4358144);   // 2048*1024 bf16
  __bf16* Vb    = (__bf16*)(w + 8552448);   // 512*1024 bf16
  __bf16* ubf   = (__bf16*)(w + 9601024);   // 512*2048 bf16
  float*  fbres = (float*)(w + 11698176);   // 512*2048 f32

  k_prep<<<64, 256, 0, stream>>>(rUr, iUr, UrT, Ur, UrTb);
  k_convA<<<8192, 256, 0, stream>>>(A, Ab, Atb);
  k_V<<<2048, 256, 0, stream>>>(theta, z, l1, Vb);
  k1_channel<<<512, 256, 0, stream>>>(sigma2, X, XXp, Y, YYp, H, lamW, rUt, iUt,
                                      UrT, Ur, UrTb, eps, tao, factor, outH,
                                      outLam, fbres);
  k2_u<<<512, 256, 0, stream>>>(Atb, Vb, fbres, LamA, lamW, factor, rho, kappa,
                                outU, ubf);
  k3_z<<<256, 256, 0, stream>>>(Ab, ubf, theta, z, l1, relax, accP, outZ, outL);
  k4_x<<<4608, 256, 0, stream>>>(outU, rXp, iXp, accNew, outX, outXXp, outAcc);

  (void)in_sizes; (void)n_in; (void)out_size; (void)ws_size;
}
